// embedding_conv_77077483094351
// MI455X (gfx1250) — matrix operand order repaired and checked by reading, not compiled, not run on hardware
//
#include <hip/hip_runtime.h>
#include <cstddef>
#include <cstdint>

// Problem constants (reference: N=100000, E=1024, D=128)
#define N_NODES   100000
#define E_EDGES   1024
#define D_DIM     128
#define NCHUNK    64
#define CH_STEPS  49
#define CHUNK_N   (CH_STEPS * 32)            // 1568
#define NPAD      (NCHUNK * CHUNK_N)         // 100352 (multiple of 32, >= N)

#define SUMS_FLOATS (E_EDGES * D_DIM)        // 131072
#define CNT_FLOATS  (E_EDGES)                // 1024
#define HI_OFF_B    ((size_t)(SUMS_FLOATS + CNT_FLOATS) * 4)
#define PLANE_B     ((size_t)D_DIM * NPAD * 2)
#define LO_OFF_B    (HI_OFF_B + PLANE_B)
#define PART_OFF_B  (LO_OFF_B + PLANE_B)     // 32*128 f32 partial maxima
#define RBLKS       32                       // reduce-stage-1 blocks (32 e-rows each)

typedef __attribute__((ext_vector_type(16))) _Float16 v16h;
typedef __attribute__((ext_vector_type(8)))  float    v8f;
typedef unsigned short u16;

union Frag16 { v16h v; uint4 q[2]; };

// ---- CDNA5 async global->LDS copy (ASYNCcnt-tracked, no VGPR staging) ----
__device__ __forceinline__ void async_load_b128(unsigned lds_off, const void* g) {
    asm volatile("global_load_async_to_lds_b128 %0, %1, off"
                 :: "v"(lds_off), "v"(g)
                 : "memory");
}
__device__ __forceinline__ void wait_asynccnt0() {
    asm volatile("s_wait_asynccnt 0x0" ::: "memory");
}
__device__ __forceinline__ unsigned lds_off32(const void* p) {
    // generic LDS address: low 32 bits are the LDS byte offset
    return (unsigned)(uintptr_t)p;
}

// ---------------------------------------------------------------- zero init
__global__ void zero_kernel(float* __restrict__ p) {
    p[blockIdx.x * 256 + threadIdx.x] = 0.0f;
}

// ------------------------------------------- emb -> k-major f16 hi/lo planes
__global__ __launch_bounds__(256) void prep_kernel(const float* __restrict__ emb,
                                                   u16* __restrict__ ehi,
                                                   u16* __restrict__ elo) {
    __shared__ float ldsT[D_DIM][65];   // [d][n_local], pad 65 -> conflict-free
    const int tid  = threadIdx.x;
    const int lane = tid & 31;
    const int n0   = blockIdx.x * 64;

    for (int pass = 0; pass < 8; ++pass) {
        const int row = pass * 8 + (tid >> 5);
        const int n   = n0 + row;
        #pragma unroll
        for (int c = 0; c < 4; ++c) {
            const int d = lane + 32 * c;
            float v = 0.0f;
            if (n < N_NODES) v = emb[(size_t)n * D_DIM + d];
            ldsT[d][row] = v;
        }
    }
    __syncthreads();

    const int d    = tid >> 1;
    const int base = (tid & 1) * 32;
    union { u16 s[32]; uint4 q[4]; } Hh, Ll;
    #pragma unroll
    for (int i = 0; i < 32; ++i) {
        const float x = ldsT[d][base + i];
        const _Float16 h = (_Float16)x;
        const _Float16 l = (_Float16)(x - (float)h);
        Hh.s[i] = __builtin_bit_cast(u16, h);
        Ll.s[i] = __builtin_bit_cast(u16, l);
    }
    const size_t dst = (size_t)d * NPAD + n0 + base;
    uint4* gh = (uint4*)(ehi + dst);
    uint4* gl = (uint4*)(elo + dst);
    #pragma unroll
    for (int b = 0; b < 4; ++b) { gh[b] = Hh.q[b]; gl[b] = Ll.q[b]; }
}

// --------------------------------------------------------- main WMMA GEMM
// grid = (E/64, NCHUNK); block = 256 (8 waves).
// Double-buffered LDS, software-pipelined H loads, async emb-tile loads.
__global__ __launch_bounds__(256) void gemm_kernel(const int* __restrict__ Hg,
                                                   const u16* __restrict__ ehi,
                                                   const u16* __restrict__ elo,
                                                   float* __restrict__ sums,
                                                   float* __restrict__ counts) {
    __shared__ u16 mT [2][64  * 40];   // maskT[e][k], stride 40 shorts (80B)
    __shared__ u16 bHi[2][128 * 40];   // embT hi tile [d][k]
    __shared__ u16 bLo[2][128 * 40];   // embT lo tile [d][k]
    __shared__ float cbuf[16][64];

    const int tid   = threadIdx.x;
    const int lane  = tid & 31;
    const int wave  = tid >> 5;
    const int e0    = blockIdx.x * 64;
    const int nbase = blockIdx.y * CHUNK_N;
    const int e_s   = wave & 3;
    const int dh    = wave >> 2;

    // mask-loader mapping: rows r0, r0+16; 4 consecutive e-columns
    const int r0  = tid >> 4;
    const int ec4 = (tid & 15) * 4;
    // B-loader mapping: 2 slots of 16B per thread per plane
    const int sd0 = (tid * 2) >> 2,     sk0 = ((tid * 2)     & 3) * 8;
    const int sd1 = (tid * 2 + 1) >> 2, sk1 = ((tid * 2 + 1) & 3) * 8;

    const int ml   = lane & 15;
    const int half = lane >> 4;

    v8f acc[4] = {v8f{}, v8f{}, v8f{}, v8f{}};
    float cnt0 = 0.f, cnt1 = 0.f, cnt2 = 0.f, cnt3 = 0.f;

    // issue async emb-tile copies for K step starting at k0 into buffer buf
    auto issueB = [&](int k0, int buf) {
        const size_t s0 = (size_t)sd0 * NPAD + k0 + sk0;
        const size_t s1 = (size_t)sd1 * NPAD + k0 + sk1;
        async_load_b128(lds_off32(&bHi[buf][sd0 * 40 + sk0]), ehi + s0);
        async_load_b128(lds_off32(&bLo[buf][sd0 * 40 + sk0]), elo + s0);
        async_load_b128(lds_off32(&bHi[buf][sd1 * 40 + sk1]), ehi + s1);
        async_load_b128(lds_off32(&bLo[buf][sd1 * 40 + sk1]), elo + s1);
    };

    // issue H global loads for K step k0 into registers (consumed by storeM)
    auto loadH = [&](int k0, int4& h0, int4& h1) {
        h0 = make_int4(0, 0, 0, 0);
        h1 = make_int4(0, 0, 0, 0);
        const int n0 = k0 + r0;
        const int n1 = k0 + r0 + 16;
        if (n0 < N_NODES) {
            h0 = *(const int4*)&Hg[(size_t)n0 * E_EDGES + e0 + ec4];
            if (n0 + 32 < N_NODES)   // prefetch two steps ahead of consumption
                __builtin_prefetch(&Hg[(size_t)(n0 + 32) * E_EDGES + e0 + ec4], 0, 1);
        }
        if (n1 < N_NODES) {
            h1 = *(const int4*)&Hg[(size_t)n1 * E_EDGES + e0 + ec4];
            if (n1 + 32 < N_NODES)
                __builtin_prefetch(&Hg[(size_t)(n1 + 32) * E_EDGES + e0 + ec4], 0, 1);
        }
    };

    // convert to f16 {0,1}, store k-major into mask buffer, accumulate counts
    auto storeM = [&](int buf, const int4& h0, const int4& h1) {
        u16* m = mT[buf];
        m[(ec4 + 0) * 40 + r0]      = h0.x > 0 ? (u16)0x3C00 : (u16)0;
        m[(ec4 + 1) * 40 + r0]      = h0.y > 0 ? (u16)0x3C00 : (u16)0;
        m[(ec4 + 2) * 40 + r0]      = h0.z > 0 ? (u16)0x3C00 : (u16)0;
        m[(ec4 + 3) * 40 + r0]      = h0.w > 0 ? (u16)0x3C00 : (u16)0;
        m[(ec4 + 0) * 40 + r0 + 16] = h1.x > 0 ? (u16)0x3C00 : (u16)0;
        m[(ec4 + 1) * 40 + r0 + 16] = h1.y > 0 ? (u16)0x3C00 : (u16)0;
        m[(ec4 + 2) * 40 + r0 + 16] = h1.z > 0 ? (u16)0x3C00 : (u16)0;
        m[(ec4 + 3) * 40 + r0 + 16] = h1.w > 0 ? (u16)0x3C00 : (u16)0;
        cnt0 += (float)(h0.x > 0) + (float)(h1.x > 0);
        cnt1 += (float)(h0.y > 0) + (float)(h1.y > 0);
        cnt2 += (float)(h0.z > 0) + (float)(h1.z > 0);
        cnt3 += (float)(h0.w > 0) + (float)(h1.w > 0);
    };

    // ---- prologue: fill buffer 0 for step 0
    int4 h0, h1;
    issueB(nbase, 0);
    loadH(nbase, h0, h1);
    storeM(0, h0, h1);
    wait_asynccnt0();
    __syncthreads();

    // ---- pipelined main loop
    for (int s = 0; s < CH_STEPS; ++s) {
        const int  p    = s & 1;
        const bool more = (s + 1) < CH_STEPS;
        if (more) {
            issueB(nbase + (s + 1) * 32, p ^ 1);
            loadH(nbase + (s + 1) * 32, h0, h1);
        }

        // A fragment (16x32 f16): lane holds row m=ml, K runs {8h..}, {16+8h..}
        Frag16 A;
        const u16* ap = &mT[p][(e_s * 16 + ml) * 40];
        A.q[0] = *(const uint4*)&ap[8 * half];
        A.q[1] = *(const uint4*)&ap[16 + 8 * half];

        const int kb = 8 * half;    // B: lane holds col n=ml, K runs {kb..kb+7}, {16+kb..}
        #pragma unroll
        for (int dt = 0; dt < 4; ++dt) {
            const int dl = dh * 64 + dt * 16 + ml;
            Frag16 Bh, Bl;
            Bh.q[0] = *(const uint4*)&bHi[p][dl * 40 + kb];
            Bh.q[1] = *(const uint4*)&bHi[p][dl * 40 + kb + 16];
            Bl.q[0] = *(const uint4*)&bLo[p][dl * 40 + kb];
            Bl.q[1] = *(const uint4*)&bLo[p][dl * 40 + kb + 16];
            acc[dt] = __builtin_amdgcn_wmma_f32_16x16x32_f16(
                false, A.v, false, Bh.v, (short)0, acc[dt], false, false);
            acc[dt] = __builtin_amdgcn_wmma_f32_16x16x32_f16(
                false, A.v, false, Bl.v, (short)0, acc[dt], false, false);
        }

        if (more) storeM(p ^ 1, h0, h1);
        wait_asynccnt0();
        __syncthreads();
    }

    // ---- merge partial sums (C layout: vgpr r -> M = r + 8*half, lane%16 -> N)
    const int eg = e0 + e_s * 16 + 8 * half;
    const int dg = dh * 64 + ml;
    #pragma unroll
    for (int dt = 0; dt < 4; ++dt) {
        #pragma unroll
        for (int r = 0; r < 8; ++r) {
            unsafeAtomicAdd(&sums[(size_t)(eg + r) * D_DIM + dg + dt * 16], acc[dt][r]);
        }
    }

    // ---- merge counts
    __syncthreads();
    cbuf[tid >> 4][(tid & 15) * 4 + 0] = cnt0;
    cbuf[tid >> 4][(tid & 15) * 4 + 1] = cnt1;
    cbuf[tid >> 4][(tid & 15) * 4 + 2] = cnt2;
    cbuf[tid >> 4][(tid & 15) * 4 + 3] = cnt3;
    __syncthreads();
    if (tid < 64) {
        float s = 0.f;
        #pragma unroll
        for (int g = 0; g < 16; ++g) s += cbuf[g][tid];
        unsafeAtomicAdd(&counts[e0 + tid], s);
    }
}

// ------------------------------------------- means + max-reduce (2 stages)
__global__ void reduce1_kernel(const float* __restrict__ sums,
                               const float* __restrict__ counts,
                               float* __restrict__ partial) {
    const int d  = threadIdx.x;            // 128
    const int b  = blockIdx.x;             // RBLKS
    const int eb = b * (E_EDGES / RBLKS);
    float m = -__builtin_inff();
    for (int e = eb; e < eb + E_EDGES / RBLKS; ++e) {
        m = fmaxf(m, sums[e * D_DIM + d] / counts[e]);
    }
    partial[b * D_DIM + d] = m;
}

__global__ void reduce2_kernel(const float* __restrict__ partial,
                               float* __restrict__ out) {
    const int d = threadIdx.x;
    float m = -__builtin_inff();
    #pragma unroll
    for (int b = 0; b < RBLKS; ++b) m = fmaxf(m, partial[b * D_DIM + d]);
    out[d] = m;
}

// ---------------------------------------------------------------- launcher
extern "C" void kernel_launch(void* const* d_in, const int* in_sizes, int n_in,
                              void* d_out, int out_size, void* d_ws, size_t ws_size,
                              hipStream_t stream) {
    const float* emb = (const float*)d_in[0];   // [N, D] f32
    const int*   Hg  = (const int*)d_in[1];     // [N, E] i32

    char* ws       = (char*)d_ws;
    float* sums    = (float*)ws;                             // [E][D] f32
    float* counts  = (float*)(ws + (size_t)SUMS_FLOATS * 4); // [E] f32
    u16*   ehi     = (u16*)(ws + HI_OFF_B);                  // [D][NPAD] f16 hi
    u16*   elo     = (u16*)(ws + LO_OFF_B);                  // [D][NPAD] f16 lo
    float* partial = (float*)(ws + PART_OFF_B);              // [RBLKS][D] f32

    zero_kernel<<<(SUMS_FLOATS + CNT_FLOATS) / 256, 256, 0, stream>>>(sums);
    prep_kernel<<<NPAD / 64, 256, 0, stream>>>(emb, ehi, elo);
    gemm_kernel<<<dim3(E_EDGES / 64, NCHUNK), 256, 0, stream>>>(Hg, ehi, elo, sums, counts);
    reduce1_kernel<<<RBLKS, D_DIM, 0, stream>>>(sums, counts, partial);
    reduce2_kernel<<<1, D_DIM, 0, stream>>>(partial, (float*)d_out);
}
